// QKVAttention_70222715289793
// MI455X (gfx1250) — matrix operand order repaired and checked by reading, not compiled, not run on hardware
//
#include <hip/hip_runtime.h>

// CDNA5 (gfx1250) flash-attention for QKVAttention reference.
// qkv: [4, 1, 3*8*64, 2048] f32, channel-major per head; out: [4, 512, 2048] f32.
// v_wmma_f32_16x16x32_f16 for both Q^T*K and P*V^T; all fragments built from
// ds_load_b128 pairs (LDS layouts match the wave32 WMMA operand maps).
// Softmax runs in base-2 (log2e folded into the Q scale) so exp == v_exp_f32.

typedef __attribute__((ext_vector_type(16))) _Float16 v16h;
typedef __attribute__((ext_vector_type(8)))  _Float16 v8h;
typedef __attribute__((ext_vector_type(4)))  _Float16 v4h;
typedef __attribute__((ext_vector_type(8)))  float    v8f;

#define HEADS   8
#define CH      64
#define TLEN    2048
#define BT      128          // query rows per workgroup (8 waves x 16 rows)
#define BS      128          // key/value tile width
#define NSB     (TLEN / BS)  // 16 s-blocks
#define QT_STRIDE 72         // halves per row of sQT/sKT (64c + pad), 144B (16B aligned)
#define V_STRIDE  136        // halves per row of sV (128s + pad), 272B (16B aligned)
#define P_STRIDE  136        // halves per row of sP, 272B (16B aligned)
#define O_STRIDE  129        // floats per row in output staging

#define QSCALE (0.125f * 1.44269504088896f)   // 1/sqrt(64) * log2(e)
#define KSCALE 0.125f                          // 1/sqrt(64)

// two contiguous 8-half runs separated by `gap` halves -> one 16-half fragment
static __device__ __forceinline__ v16h load_frag(const _Float16* p, int gap)
{
    v8h lo = *(const v8h*)p;
    v8h hi = *(const v8h*)(p + gap);
    return __builtin_shufflevector(lo, hi, 0, 1, 2, 3, 4, 5, 6, 7,
                                           8, 9, 10, 11, 12, 13, 14, 15);
}

__global__ __launch_bounds__(256, 1)
void qkv_attn_wmma(const float* __restrict__ qkv, float* __restrict__ out)
{
    // sQT: 128t x 72   sKT: 128s x 72   sV: 64c x 136   sP: 8 waves x 16 x 136
    __shared__ __align__(32) unsigned char
        smem_raw[(BT * QT_STRIDE + BS * QT_STRIDE + CH * V_STRIDE + 8 * 16 * P_STRIDE) * 2];
    _Float16* sQT = (_Float16*)smem_raw;
    _Float16* sKT = sQT + BT * QT_STRIDE;
    _Float16* sV  = sKT + BS * QT_STRIDE;
    _Float16* sP  = sV + CH * V_STRIDE;

    const int tid  = threadIdx.x;
    const int wave = tid >> 5;
    const int lane = tid & 31;
    const int half = lane >> 4;       // which 16-lane half
    const int colN = lane & 15;       // B / C-D column
    const int rowM = lane & 15;       // A row
    const int kHalf   = half * 8;     // A-layout K split
    const int kBase16 = half * 16;    // contiguous K split (unused)

    const int b  = blockIdx.y >> 3;
    const int h  = blockIdx.y & 7;
    const int t0 = blockIdx.x * BT;

    const float* qbase = qkv + ((size_t)(b * 3 * HEADS * CH) + h * CH) * (size_t)TLEN;
    const float* kbase = qbase + (size_t)(HEADS * CH) * TLEN;
    const float* vbase = qbase + (size_t)(2 * HEADS * CH) * TLEN;

    // ---- Stage Q tile transposed: sQT[t][c], pre-scaled by 0.125*log2e ----
    #pragma unroll
    for (int i = 0; i < 8; ++i) {
        int linear = tid * 4 + i * 1024;       // multiples of 4 covering 0..8191
        int c = linear >> 7;
        int t = linear & 127;
        float4 v = *(const float4*)(qbase + (size_t)c * TLEN + t0 + t);
        sQT[(t + 0) * QT_STRIDE + c] = (_Float16)(v.x * QSCALE);
        sQT[(t + 1) * QT_STRIDE + c] = (_Float16)(v.y * QSCALE);
        sQT[(t + 2) * QT_STRIDE + c] = (_Float16)(v.z * QSCALE);
        sQT[(t + 3) * QT_STRIDE + c] = (_Float16)(v.w * QSCALE);
    }
    __syncthreads();

    // ---- Q^T A-fragments (16 t-rows x 64 c): 2x ds_load_b128 each ----
    v16h aQ[2];
    #pragma unroll
    for (int cc = 0; cc < 2; ++cc)
        aQ[cc] = load_frag(sQT + (wave * 16 + rowM) * QT_STRIDE + cc * 32 + kHalf, 16);

    // ---- Flash-attention running state (C/D layout: VGPR r -> row r+8*half) ----
    // rowMax is in base-2 (log2e-scaled) units throughout.
    float rowMax[8], rowSum[8];
    v8f accO[4];
    #pragma unroll
    for (int r = 0; r < 8; ++r) { rowMax[r] = -INFINITY; rowSum[r] = 0.0f; }
    #pragma unroll
    for (int j = 0; j < 4; ++j)
        #pragma unroll
        for (int r = 0; r < 8; ++r) accO[j][r] = 0.0f;

    _Float16* sPw = sP + wave * 16 * P_STRIDE;

    for (int sb = 0; sb < NSB; ++sb) {
        const int s0 = sb * BS;
        __syncthreads();   // previous K/V tile fully consumed

        // ---- Stage K transposed (scaled) and V (natural) as f16 ----
        #pragma unroll
        for (int i = 0; i < 8; ++i) {
            int linear = tid * 4 + i * 1024;
            int c = linear >> 7;
            int t = linear & 127;
            float4 kv = *(const float4*)(kbase + (size_t)c * TLEN + s0 + t);
            float4 vv = *(const float4*)(vbase + (size_t)c * TLEN + s0 + t);
            sKT[(t + 0) * QT_STRIDE + c] = (_Float16)(kv.x * KSCALE);
            sKT[(t + 1) * QT_STRIDE + c] = (_Float16)(kv.y * KSCALE);
            sKT[(t + 2) * QT_STRIDE + c] = (_Float16)(kv.z * KSCALE);
            sKT[(t + 3) * QT_STRIDE + c] = (_Float16)(kv.w * KSCALE);
            v4h vpk;
            vpk[0] = (_Float16)vv.x; vpk[1] = (_Float16)vv.y;
            vpk[2] = (_Float16)vv.z; vpk[3] = (_Float16)vv.w;
            *(v4h*)(sV + c * V_STRIDE + t) = vpk;   // one ds_store_b64
        }
        __syncthreads();

        // Prefetch next s-block of K/V into L2 (global_prefetch_b8)
        if (sb + 1 < NSB) {
            int s0n = s0 + BS;
            int pc  = tid & 63;
            int seg = (tid >> 6) * 32;
            __builtin_prefetch(kbase + (size_t)pc * TLEN + s0n + seg, 0, 1);
            __builtin_prefetch(vbase + (size_t)pc * TLEN + s0n + seg, 0, 1);
        }

        // ---- Scores (base-2 units): S[16x128]; A-stationary over 8 N-tiles ----
        v8f sc[8];
        #pragma unroll
        for (int j = 0; j < 8; ++j)
            #pragma unroll
            for (int r = 0; r < 8; ++r) sc[j][r] = 0.0f;
        #pragma unroll
        for (int cc = 0; cc < 2; ++cc) {
            #pragma unroll
            for (int j = 0; j < 8; ++j) {
                v16h bK = load_frag(sKT + (j * 16 + colN) * QT_STRIDE + cc * 32 + kHalf, 16);
                sc[j] = __builtin_amdgcn_wmma_f32_16x16x32_f16(
                    false, aQ[cc], false, bK, (short)0, sc[j], false, false);
            }
        }

        // ---- Online softmax update, base-2 (row reductions over 16-lane half) ----
        #pragma unroll
        for (int r = 0; r < 8; ++r) {
            float m = sc[0][r];
            #pragma unroll
            for (int j = 1; j < 8; ++j) m = fmaxf(m, sc[j][r]);
            m = fmaxf(m, __shfl_xor(m, 1, 32));
            m = fmaxf(m, __shfl_xor(m, 2, 32));
            m = fmaxf(m, __shfl_xor(m, 4, 32));
            m = fmaxf(m, __shfl_xor(m, 8, 32));
            float nmax  = fmaxf(rowMax[r], m);
            float alpha = exp2f(rowMax[r] - nmax);   // bare v_exp_f32
            rowMax[r] = nmax;

            float ls = 0.0f;
            #pragma unroll
            for (int j = 0; j < 8; ++j) {
                float p = exp2f(sc[j][r] - nmax);    // bare v_exp_f32
                sc[j][r] = p;
                ls += p;
            }
            ls += __shfl_xor(ls, 1, 32);
            ls += __shfl_xor(ls, 2, 32);
            ls += __shfl_xor(ls, 4, 32);
            ls += __shfl_xor(ls, 8, 32);
            rowSum[r] = rowSum[r] * alpha + ls;
            #pragma unroll
            for (int j = 0; j < 4; ++j) accO[j][r] *= alpha;
        }

        // ---- C/D layout -> A layout for P via per-wave LDS bounce (f16) ----
        #pragma unroll
        for (int j = 0; j < 8; ++j)
            #pragma unroll
            for (int r = 0; r < 8; ++r)
                sPw[(r + 8 * half) * P_STRIDE + j * 16 + colN] = (_Float16)sc[j][r];

        // ---- O += P * V^T : 4 s-chunks of 32 x 4 c-tiles of 16, A-stationary ----
        #pragma unroll
        for (int sk = 0; sk < 4; ++sk) {
            v16h aP = load_frag(sPw + rowM * P_STRIDE + sk * 32 + kHalf, 16);
            #pragma unroll
            for (int j = 0; j < 4; ++j) {
                v16h bV = load_frag(sV + (j * 16 + colN) * V_STRIDE + sk * 32 + kHalf, 16);
                accO[j] = __builtin_amdgcn_wmma_f32_16x16x32_f16(
                    false, aP, false, bV, (short)0, accO[j], false, false);
            }
        }
    }

    // ---- Normalize (1 rcp per row), transpose through LDS, coalesced stores ----
    __syncthreads();
    float* sO = (float*)smem_raw;  // 64c x O_STRIDE floats (reuses sQT/sKT region)
    #pragma unroll
    for (int r = 0; r < 8; ++r) {
        float inv = __frcp_rn(rowSum[r]);
        #pragma unroll
        for (int j = 0; j < 4; ++j)
            sO[(j * 16 + colN) * O_STRIDE + wave * 16 + r + 8 * half] =
                accO[j][r] * inv;
    }
    __syncthreads();

    float* obase = out + ((size_t)(b * HEADS * CH) + h * CH) * (size_t)TLEN + t0;
    #pragma unroll
    for (int i = 0; i < 8; ++i) {
        int linear = tid * 4 + i * 1024;
        int c = linear >> 7;
        int t = linear & 127;
        const float* src = sO + c * O_STRIDE + t;
        float4 v;
        v.x = src[0]; v.y = src[1]; v.z = src[2]; v.w = src[3];
        *(float4*)(obase + (size_t)c * TLEN + t) = v;
    }
}

extern "C" void kernel_launch(void* const* d_in, const int* in_sizes, int n_in,
                              void* d_out, int out_size, void* d_ws, size_t ws_size,
                              hipStream_t stream)
{
    (void)in_sizes; (void)n_in; (void)out_size; (void)d_ws; (void)ws_size;
    const float* qkv = (const float*)d_in[0];
    float* out = (float*)d_out;
    dim3 grid(TLEN / BT, 4 * HEADS);   // 16 query tiles x 32 (b,h) pairs
    dim3 block(256);
    qkv_attn_wmma<<<grid, block, 0, stream>>>(qkv, out);
}
